// lstm_target_28879360098508
// MI455X (gfx1250) — matrix operand order repaired and checked by reading, not compiled, not run on hardware
//
#include <hip/hip_runtime.h>

// ---------------------------------------------------------------------------
// B=256, S=20, E=512, H=512, V=32000, C=1000
// d_in1 = S*E + 2E = 11264 ; K1 = d_in1 + H = 11776 ; 4H = 2048
// ---------------------------------------------------------------------------

typedef __attribute__((ext_vector_type(16))) __bf16 v16bf;
typedef __attribute__((ext_vector_type(8)))  __bf16 v8bf;
typedef __attribute__((ext_vector_type(2)))  __bf16 v2bf;
typedef __attribute__((ext_vector_type(8)))  float  v8f;

__device__ __forceinline__ unsigned short f2bf(float f) {
    __bf16 h = (__bf16)f;                        // v_cvt_pk_bf16_f32 (RNE)
    return __builtin_bit_cast(unsigned short, h);
}
// pack two f32 -> one dword of bf16 pair; vector build lets ISel match
// a single v_cvt_pk_bf16_f32 with both operands.
__device__ __forceinline__ unsigned int pack2bf(float lo, float hi) {
    v2bf p;
    p.x = (__bf16)lo;
    p.y = (__bf16)hi;
    return __builtin_bit_cast(unsigned int, p);
}

// -------------------- input gather: xh = [enc | wemb | persona | h0] (bf16) --
__global__ void __launch_bounds__(256)
build_xh(const float* __restrict__ enc, const float* __restrict__ wemb,
         const float* __restrict__ h0, const int* __restrict__ speaker,
         const float* __restrict__ persona, unsigned short* __restrict__ xh)
{
    const int D = 11776;
    int idx = blockIdx.x * 256 + threadIdx.x;          // grid = 11776 blocks
    int b = idx / D;
    int c = idx - b * D;
    float v;
    if (c < 10240)       v = enc[(size_t)b * 10240 + c];
    else if (c < 10752)  v = wemb[(size_t)b * 512 + (c - 10240)];
    else if (c < 11264)  v = persona[(size_t)speaker[b] * 512 + (c - 10752)];
    else                 v = h0[(size_t)b * 512 + (c - 11264)];
    xh[idx] = f2bf(v);
}

// -------------------- bf16 WMMA GEMM:  Out(MxN) = A(MxK) @ [B0;B1] + bias ----
// A: bf16 row-major, stride K.  B0/B1: fp32 row-major (K x N), split at ksplit
// (ksplit MUST be a multiple of 32 -- true for all call sites: 11264, 512).
// Block tile 128x128x32; 256 threads = 8 waves in 4(M)x2(N);
// wave tile 32x64 = 2x4 subtiles = 8 v_wmma per wave per k-step.
// 2-stage software pipeline: global loads for k+32 overlap WMMA on k.
__global__ void __launch_bounds__(256)
gemm_bf16_wmma(const unsigned short* __restrict__ A,
               const float* __restrict__ B0, const float* __restrict__ B1,
               int ksplit, const float* __restrict__ bias,
               float* __restrict__ Out, int N, int K)
{
    __shared__ __align__(32) unsigned short sA[128 * 32];  // [row][k] bf16
    __shared__ __align__(32) unsigned short sB[128 * 32];  // [col][k] bf16 (N-major)

    const int tid  = threadIdx.x;
    const int bm   = blockIdx.y * 128;
    const int bn   = blockIdx.x * 128;
    const int lane = tid & 31, wave = tid >> 5;
    const int wm   = wave >> 1, wn = wave & 1;             // 4 x 2 waves
    const int l15  = lane & 15, hi = lane >> 4;

    const unsigned int* A32 = (const unsigned int*)A;
    unsigned int* sA32 = (unsigned int*)sA;
    unsigned int* sB32 = (unsigned int*)sB;

    v8f acc[2][4] = {};
    unsigned int ra[8];
    float rb[16];

    auto ld_stage = [&](int k0) {
        #pragma unroll
        for (int i = 0; i < 8; ++i) {                      // A: 128x32 bf16 = 2048 dwords
            int e = tid + i * 256;
            int r = e >> 4, cp = e & 15;
            ra[i] = A32[(((size_t)(bm + r) * K + k0) >> 1) + cp];
        }
        // whole 32-deep slab is on one side of ksplit (ksplit % 32 == 0)
        const float* Bk = (k0 < ksplit)
            ? (B0 + (size_t)k0 * N + bn)
            : (B1 + (size_t)(k0 - ksplit) * N + bn);
        #pragma unroll
        for (int i = 0; i < 8; ++i) {                      // B: 128n x 32k fp32, pair in k
            int e = tid + i * 256;
            int n = e & 127, kp = e >> 7;                  // kp = 0..15
            const float* s0 = Bk + (size_t)(2 * kp) * N + n;
            rb[2 * i]     = s0[0];
            rb[2 * i + 1] = s0[N];
        }
    };
    auto st_stage = [&]() {
        #pragma unroll
        for (int i = 0; i < 8; ++i)
            sA32[tid + i * 256] = ra[i];
        #pragma unroll
        for (int i = 0; i < 8; ++i) {
            int e = tid + i * 256;
            int n = e & 127, kp = e >> 7;
            sB32[n * 16 + kp] = pack2bf(rb[2 * i], rb[2 * i + 1]);
        }
    };

    ld_stage(0);
    st_stage();

    for (int k0 = 0; k0 < K; k0 += 32) {
        __syncthreads();                                   // LDS tile ready
        const bool more = (k0 + 32) < K;
        if (more) ld_stage(k0 + 32);                       // overlap with WMMA below

        v16bf afr[2], bfr[4];
        #pragma unroll
        for (int sm = 0; sm < 2; ++sm) {
            int row = wm * 32 + sm * 16 + l15;
            v8bf alo = *reinterpret_cast<const v8bf*>(sA + row * 32 + hi * 16);
            v8bf ahi = *reinterpret_cast<const v8bf*>(sA + row * 32 + hi * 16 + 8);
            afr[sm] = __builtin_shufflevector(alo, ahi,
                      0,1,2,3,4,5,6,7,8,9,10,11,12,13,14,15);
        }
        #pragma unroll
        for (int sn = 0; sn < 4; ++sn) {
            int col = wn * 64 + sn * 16 + l15;
            bfr[sn] = *reinterpret_cast<const v16bf*>(sB + col * 32 + hi * 16);
        }
        #pragma unroll
        for (int sm = 0; sm < 2; ++sm)
            #pragma unroll
            for (int sn = 0; sn < 4; ++sn)
                acc[sm][sn] = __builtin_amdgcn_wmma_f32_16x16x32_bf16(
                    false, afr[sm], false, bfr[sn],
                    (short)0, acc[sm][sn], false, false);

        __syncthreads();                                   // all waves done reading
        if (more) st_stage();
    }

    // epilogue: C/D layout -> lane l<16:(M=r,N=l), l>=16:(M=r+8,N=l-16)
    #pragma unroll
    for (int sn = 0; sn < 4; ++sn) {
        int nn = bn + wn * 64 + sn * 16 + l15;
        float bv = bias[nn];
        #pragma unroll
        for (int sm = 0; sm < 2; ++sm) {
            #pragma unroll
            for (int r = 0; r < 8; ++r) {
                int m = bm + wm * 32 + sm * 16 + hi * 8 + r;
                Out[(size_t)m * N + nn] = acc[sm][sn][r] + bv;
            }
        }
    }
}

// -------------------- LSTM pointwise cell -----------------------------------
__global__ void __launch_bounds__(256)
lstm_cell(const float* __restrict__ z, const float* __restrict__ c_prev,
          float* __restrict__ c_out, unsigned short* __restrict__ h_bf,
          int h_stride, int dup, float* __restrict__ h_f32)
{
    int idx = blockIdx.x * 256 + threadIdx.x;          // 256*512 threads
    int b = idx >> 9, hc = idx & 511;
    const float* zb = z + (size_t)b * 2048;
    float zi = zb[hc], zf = zb[512 + hc], zg = zb[1024 + hc], zo = zb[1536 + hc];
    float ig = 1.f / (1.f + __expf(-zi));
    float fg = 1.f / (1.f + __expf(-zf));
    float gg = fmaxf(zg, 0.f);
    float og = 1.f / (1.f + __expf(-zo));
    float c  = fg * c_prev[idx] + ig * gg;
    float h  = og * fmaxf(c, 0.f);
    c_out[idx] = c;
    unsigned short hb = f2bf(h);
    h_bf[(size_t)b * h_stride + hc] = hb;
    if (dup) h_bf[(size_t)b * h_stride + 512 + hc] = hb;
    if (h_f32) h_f32[idx] = h;
}

// -------------------- row softmax (V = 32000) --------------------------------
__global__ void __launch_bounds__(256)
softmax_rows(const float* __restrict__ logits, float* __restrict__ probs)
{
    __shared__ float red[256];
    const int b = blockIdx.x, tid = threadIdx.x;
    const float* row = logits + (size_t)b * 32000;
    float m = -3.402823466e38f;
    for (int j = tid; j < 32000; j += 256) m = fmaxf(m, row[j]);
    red[tid] = m; __syncthreads();
    for (int s = 128; s > 0; s >>= 1) {
        if (tid < s) red[tid] = fmaxf(red[tid], red[tid + s]);
        __syncthreads();
    }
    m = red[0]; __syncthreads();
    float sum = 0.f;
    for (int j = tid; j < 32000; j += 256) sum += __expf(row[j] - m);
    red[tid] = sum; __syncthreads();
    for (int s = 128; s > 0; s >>= 1) {
        if (tid < s) red[tid] += red[tid + s];
        __syncthreads();
    }
    float inv = 1.f / red[0];
    float* orow = probs + (size_t)b * 32000;
    for (int j = tid; j < 32000; j += 256) orow[j] = __expf(row[j] - m) * inv;
}

// ---------------------------------------------------------------------------
extern "C" void kernel_launch(void* const* d_in, const int* in_sizes, int n_in,
                              void* d_out, int out_size, void* d_ws, size_t ws_size,
                              hipStream_t stream)
{
    const float* enc     = (const float*)d_in[0];
    const float* wemb    = (const float*)d_in[1];
    const float* h0      = (const float*)d_in[2];
    const float* c0      = (const float*)d_in[3];
    const int*   speaker = (const int*)  d_in[4];
    /* d_in[5] addressee unused (is_speaker branch) */
    const float* persona = (const float*)d_in[6];
    const float* W1 = (const float*)d_in[7],  *U1 = (const float*)d_in[8],  *b1 = (const float*)d_in[9];
    const float* W2 = (const float*)d_in[10], *U2 = (const float*)d_in[11], *b2 = (const float*)d_in[12];
    const float* W3 = (const float*)d_in[13], *U3 = (const float*)d_in[14], *b3 = (const float*)d_in[15];
    const float* W4 = (const float*)d_in[16], *U4 = (const float*)d_in[17], *b4 = (const float*)d_in[18];
    const float* Wd = (const float*)d_in[19], *bd = (const float*)d_in[20];

    float* out   = (float*)d_out;
    float* probs = out;                         // 256*32000
    float* h4out = out + (size_t)256 * 32000;   // 256*512
    float* c4out = h4out + (size_t)256 * 512;   // 256*512

    char* ws = (char*)d_ws;
    size_t off = 0;
    unsigned short* xh   = (unsigned short*)(ws + off); off += (size_t)256 * 11776 * 2; // 6.03 MB
    unsigned short* hh   = (unsigned short*)(ws + off); off += (size_t)256 * 1024  * 2; // 0.52 MB
    float*          z    = (float*)         (ws + off); off += (size_t)256 * 2048  * 4; // 2.10 MB
    float*          cbuf = (float*)         (ws + off); off += (size_t)256 * 512   * 4; // 0.52 MB
    unsigned short* h4bf = (unsigned short*)(ws + off); off += (size_t)256 * 512   * 2; // 0.26 MB
    float*          logz = (float*)         (ws + off);                                 // 32.77 MB

    // 1) gather + bf16 convert inputs
    build_xh<<<11776, 256, 0, stream>>>(enc, wemb, h0, speaker, persona, xh);

    // 2) layer 1: z = xh @ [W1;U1] + b1   (K = 11776, split 11264)
    gemm_bf16_wmma<<<dim3(16, 2), 256, 0, stream>>>(xh, W1, U1, 11264, b1, z, 2048, 11776);
    lstm_cell<<<512, 256, 0, stream>>>(z, c0, cbuf, hh, 1024, 1, nullptr);

    // 3) layers 2..4: z = [h|h] @ [W;U] + b  (K = 1024, split 512)
    gemm_bf16_wmma<<<dim3(16, 2), 256, 0, stream>>>(hh, W2, U2, 512, b2, z, 2048, 1024);
    lstm_cell<<<512, 256, 0, stream>>>(z, cbuf, cbuf, hh, 1024, 1, nullptr);

    gemm_bf16_wmma<<<dim3(16, 2), 256, 0, stream>>>(hh, W3, U3, 512, b3, z, 2048, 1024);
    lstm_cell<<<512, 256, 0, stream>>>(z, cbuf, cbuf, hh, 1024, 1, nullptr);

    gemm_bf16_wmma<<<dim3(16, 2), 256, 0, stream>>>(hh, W4, U4, 512, b4, z, 2048, 1024);
    lstm_cell<<<512, 256, 0, stream>>>(z, cbuf, c4out, h4bf, 512, 0, h4out);

    // 4) decoder: logits = h4 @ Wd + bd  (K = 512, no split), then softmax
    gemm_bf16_wmma<<<dim3(250, 2), 256, 0, stream>>>(h4bf, Wd, Wd, 512, bd, logz, 32000, 512);
    softmax_rows<<<256, 256, 0, stream>>>(logz, probs);
}
